// EfficientAdditiveAttentions_52158082842782
// MI455X (gfx1250) — compile-verified, matrix operand order checked by reading, not run on hardware
//
#include <hip/hip_runtime.h>
#include <hip/hip_bf16.h>

// ---------- types for WMMA ----------
typedef __attribute__((ext_vector_type(16))) __bf16 v16bf;
typedef __attribute__((ext_vector_type(8)))  float  v8f;

// ---------- helpers ----------
__device__ __forceinline__ unsigned short f2bf(float f) {
    unsigned int u = __builtin_bit_cast(unsigned int, f);
    u += 0x7fffu + ((u >> 16) & 1u);   // round-to-nearest-even
    return (unsigned short)(u >> 16);
}
__device__ __forceinline__ float bf2f(unsigned short u) {
    return __builtin_bit_cast(float, ((unsigned int)u) << 16);
}
__device__ __forceinline__ float gelu_exact(float x) {
    return 0.5f * x * (1.0f + erff(x * 0.70710678118654752440f));
}
// load 16 bf16 (two 16B chunks 16 elements apart) -> one WMMA fragment
__device__ __forceinline__ v16bf ldfrag(const unsigned short* p) {
    struct alignas(32) F { uint4 a, b; } f;
    f.a = *reinterpret_cast<const uint4*>(p);
    f.b = *reinterpret_cast<const uint4*>(p + 16);
    return __builtin_bit_cast(v16bf, f);
}

// ---------- constants ----------
#define BB   16
#define CC   512
#define NN   3136        // 56*56
#define DD   512
#define TD   256
#define RT   50176       // BB*NN

// ---------- tiled transpose + fp32->bf16 convert: out[q*P+p] = bf16(in[p*Q+q]) ----------
__global__ void transpose_f32_bf16(const float* __restrict__ in, unsigned short* __restrict__ out,
                                   int P, int Q, size_t inBS, size_t outBS) {
    __shared__ float tile[32][33];
    const float* ip = in + (size_t)blockIdx.z * inBS;
    unsigned short* op = out + (size_t)blockIdx.z * outBS;
    int q0 = blockIdx.x * 32, p0 = blockIdx.y * 32;
    int tx = threadIdx.x, ty = threadIdx.y;        // 32 x 8
#pragma unroll
    for (int j = 0; j < 32; j += 8)
        tile[ty + j][tx] = ip[(size_t)(p0 + ty + j) * Q + q0 + tx];
    __syncthreads();
#pragma unroll
    for (int j = 0; j < 32; j += 8)
        op[(size_t)(q0 + ty + j) * P + p0 + tx] = f2bf(tile[tx][ty + j]);
}

// ---------- epilogue for one 16x16 accumulator tile ----------
// mode 0: Cf[m*N+n]  = acc + bias[n]
// mode 1: Cbf[m*N+n] = bf16(gelu(acc + bias[n]))
// mode 2: Cf[(b*256 + m)*3136 + n%3136] = acc + bias[m]   (transposed store, rows=channel)
__device__ __forceinline__ void epilogue_tile(v8f acc, int mb, int nb, int N, int mode,
                                              const float* __restrict__ bias,
                                              float* __restrict__ Cf,
                                              unsigned short* __restrict__ Cbf,
                                              int r, int half) {
    if (mode == 0) {
        float bv = bias[nb + r];
#pragma unroll
        for (int v = 0; v < 8; ++v)
            Cf[(size_t)(mb + v + 8 * half) * N + nb + r] = acc[v] + bv;
    } else if (mode == 1) {
        float bv = bias[nb + r];
#pragma unroll
        for (int v = 0; v < 8; ++v)
            Cbf[(size_t)(mb + v + 8 * half) * N + nb + r] = f2bf(gelu_exact(acc[v] + bv));
    } else {
        int ng = nb + r;
        int b = ng / NN, nn = ng - b * NN;
#pragma unroll
        for (int v = 0; v < 8; ++v) {
            int c = mb + v + 8 * half;
            Cf[((size_t)(b * TD + c)) * NN + nn] = acc[v] + bias[c];
        }
    }
}

// ---------- WMMA GEMM: D = A[M,K=512] * Bt[N,K=512]^T ----------
// 2x2 register tiling: each wave computes a 32x32 output block (4 accumulators),
// 4 fragment loads -> 4 WMMAs per k-step (1:1 load:wmma issue ratio).
__global__ void gemm16(const unsigned short* __restrict__ A,
                       const unsigned short* __restrict__ Bt,
                       const float* __restrict__ bias,
                       float* __restrict__ Cf, unsigned short* __restrict__ Cbf,
                       int M, int N, int mode) {
    int t = blockIdx.x * 8 + (threadIdx.x >> 5);
    int nt2 = N >> 5;                       // 32-wide supertiles
    if (t >= (M >> 5) * nt2) return;
    int sm = t / nt2, sn = t - sm * nt2;
    int m0 = sm << 5, n0 = sn << 5;

    int lane = threadIdx.x & 31;
    int r = lane & 15, half = lane >> 4;
    int off = half ? 8 : 0;

    const unsigned short* ap0 = A  + (size_t)(m0 + r) * 512 + off;
    const unsigned short* ap1 = ap0 + (size_t)16 * 512;
    const unsigned short* bp0 = Bt + (size_t)(n0 + r) * 512 + off;
    const unsigned short* bp1 = bp0 + (size_t)16 * 512;
    __builtin_prefetch(ap0, 0, 1);
    __builtin_prefetch(ap1, 0, 1);
    __builtin_prefetch(bp0, 0, 1);
    __builtin_prefetch(bp1, 0, 1);

    v8f c00 = {}, c01 = {}, c10 = {}, c11 = {};
#pragma unroll 2
    for (int k = 0; k < 512; k += 32) {
        v16bf a0 = ldfrag(ap0 + k);
        v16bf a1 = ldfrag(ap1 + k);
        v16bf b0 = ldfrag(bp0 + k);
        v16bf b1 = ldfrag(bp1 + k);
        c00 = __builtin_amdgcn_wmma_f32_16x16x32_bf16(false, a0, false, b0, (short)0, c00, false, false);
        c01 = __builtin_amdgcn_wmma_f32_16x16x32_bf16(false, a0, false, b1, (short)0, c01, false, false);
        c10 = __builtin_amdgcn_wmma_f32_16x16x32_bf16(false, a1, false, b0, (short)0, c10, false, false);
        c11 = __builtin_amdgcn_wmma_f32_16x16x32_bf16(false, a1, false, b1, (short)0, c11, false, false);
    }

    epilogue_tile(c00, m0,      n0,      N, mode, bias, Cf, Cbf, r, half);
    epilogue_tile(c01, m0,      n0 + 16, N, mode, bias, Cf, Cbf, r, half);
    epilogue_tile(c10, m0 + 16, n0,      N, mode, bias, Cf, Cbf, r, half);
    epilogue_tile(c11, m0 + 16, n0 + 16, N, mode, bias, Cf, Cbf, r, half);
}

// ---------- row L2-normalize (in place), optional bf16 copy + gated dot ----------
__global__ void normalize_rows(float* __restrict__ X, unsigned short* __restrict__ Xbf,
                               const float* __restrict__ wg, float* __restrict__ qg,
                               int R, int flags) {
    int w = (blockIdx.x * blockDim.x + threadIdx.x) >> 5;
    if (w >= R) return;
    int lane = threadIdx.x & 31;
    float* row = X + (size_t)w * 512 + lane * 16;
    float4 v[4];
    const float4* rp = reinterpret_cast<const float4*>(row);
#pragma unroll
    for (int i = 0; i < 4; ++i) v[i] = rp[i];
    float vals[16];
#pragma unroll
    for (int i = 0; i < 4; ++i) {
        vals[4*i+0] = v[i].x; vals[4*i+1] = v[i].y; vals[4*i+2] = v[i].z; vals[4*i+3] = v[i].w;
    }
    float s = 0.f;
#pragma unroll
    for (int i = 0; i < 16; ++i) s += vals[i] * vals[i];
#pragma unroll
    for (int off = 16; off; off >>= 1) s += __shfl_xor(s, off, 32);
    float inv = 1.0f / fmaxf(sqrtf(s), 1e-12f);
#pragma unroll
    for (int i = 0; i < 16; ++i) vals[i] *= inv;
    float4* wp4 = reinterpret_cast<float4*>(row);
#pragma unroll
    for (int i = 0; i < 4; ++i) {
        float4 o; o.x = vals[4*i+0]; o.y = vals[4*i+1]; o.z = vals[4*i+2]; o.w = vals[4*i+3];
        wp4[i] = o;
    }
    if (flags & 1) {
        unsigned int pk[8];
#pragma unroll
        for (int j = 0; j < 8; ++j)
            pk[j] = (unsigned int)f2bf(vals[2*j]) | ((unsigned int)f2bf(vals[2*j+1]) << 16);
        uint4* bp = reinterpret_cast<uint4*>(Xbf + (size_t)w * 512 + lane * 16);
        uint4 q0; q0.x = pk[0]; q0.y = pk[1]; q0.z = pk[2]; q0.w = pk[3];
        uint4 q1; q1.x = pk[4]; q1.y = pk[5]; q1.z = pk[6]; q1.w = pk[7];
        bp[0] = q0; bp[1] = q1;
    }
    if (flags & 2) {
        const float* wgp = wg + lane * 16;
        float d = 0.f;
#pragma unroll
        for (int i = 0; i < 16; ++i) d += vals[i] * wgp[i];
#pragma unroll
        for (int off = 16; off; off >>= 1) d += __shfl_xor(d, off, 32);
        if (lane == 0) qg[w] = d * 0.0625f;   // * 256^-0.5
    }
}

// ---------- per-batch 1/max(||qg||, eps) ----------
__global__ void batch_rnorm(const float* __restrict__ qg, float* __restrict__ rn) {
    __shared__ float red[8];
    int b = blockIdx.x;
    float s = 0.f;
    for (int n = threadIdx.x; n < NN; n += 256) {
        float v = qg[b * NN + n];
        s += v * v;
    }
#pragma unroll
    for (int off = 16; off; off >>= 1) s += __shfl_xor(s, off, 32);
    if ((threadIdx.x & 31) == 0) red[threadIdx.x >> 5] = s;
    __syncthreads();
    if (threadIdx.x == 0) {
        float t = 0.f;
        for (int i = 0; i < 8; ++i) t += red[i];
        rn[b] = 1.0f / fmaxf(sqrtf(t), 1e-12f);
    }
}

// ---------- G[b,d] = rn[b] * sum_n qg[b,n] * Q[b,n,d] ----------
__global__ void global_query(const float* __restrict__ qg, const float* __restrict__ rn,
                             const float* __restrict__ Q, float* __restrict__ G) {
    int b = blockIdx.x;
    int d = blockIdx.y * 256 + threadIdx.x;
    const float* qb = Q + ((size_t)b * NN) * 512 + d;
    const float* gb = qg + b * NN;
    float s = 0.f;
    for (int n = 0; n < NN; ++n) s += gb[n] * qb[(size_t)n * 512];
    G[b * 512 + d] = s * rn[b];
}

// ---------- dyn[row] = H[row,:] . wd2 + bd2 ----------
__global__ void dyn_rowdot(const unsigned short* __restrict__ H, const float* __restrict__ wd2,
                           const float* __restrict__ bd2, float* __restrict__ dyn, int R) {
    int w = (blockIdx.x * blockDim.x + threadIdx.x) >> 5;
    if (w >= R) return;
    int lane = threadIdx.x & 31;
    const unsigned short* hp = H + (size_t)w * 512 + lane * 16;
    uint4 u0 = *reinterpret_cast<const uint4*>(hp);
    uint4 u1 = *reinterpret_cast<const uint4*>(hp + 8);
    unsigned int uu[8] = {u0.x, u0.y, u0.z, u0.w, u1.x, u1.y, u1.z, u1.w};
    const float* wp = wd2 + lane * 16;
    float s = 0.f;
#pragma unroll
    for (int i = 0; i < 8; ++i) {
        s += bf2f((unsigned short)(uu[i] & 0xffffu)) * wp[2*i];
        s += bf2f((unsigned short)(uu[i] >> 16))     * wp[2*i+1];
    }
#pragma unroll
    for (int off = 16; off; off >>= 1) s += __shfl_xor(s, off, 32);
    if (lane == 0) dyn[w] = s + bd2[0];
}

// ---------- per-batch softmax over 3136 tokens ----------
__global__ void softmax_rows(const float* __restrict__ dyn, float* __restrict__ wsm) {
    __shared__ float red[8];
    __shared__ float bval;
    int b = blockIdx.x, tid = threadIdx.x;
    const float* din = dyn + b * NN;
    float* wout = wsm + b * NN;
    float mx = -3.402823466e38f;
    for (int n = tid; n < NN; n += 256) mx = fmaxf(mx, din[n]);
#pragma unroll
    for (int off = 16; off; off >>= 1) mx = fmaxf(mx, __shfl_xor(mx, off, 32));
    if ((tid & 31) == 0) red[tid >> 5] = mx;
    __syncthreads();
    if (tid == 0) {
        float m = red[0];
        for (int i = 1; i < 8; ++i) m = fmaxf(m, red[i]);
        bval = m;
    }
    __syncthreads();
    float m = bval, s = 0.f;
    for (int n = tid; n < NN; n += 256) {
        float e = expf(din[n] - m);
        wout[n] = e;
        s += e;
    }
#pragma unroll
    for (int off = 16; off; off >>= 1) s += __shfl_xor(s, off, 32);
    __syncthreads();
    if ((tid & 31) == 0) red[tid >> 5] = s;
    __syncthreads();
    if (tid == 0) {
        float t = 0.f;
        for (int i = 0; i < 8; ++i) t += red[i];
        bval = 1.0f / t;
    }
    __syncthreads();
    float inv = bval;
    for (int n = tid; n < NN; n += 256) wout[n] *= inv;
}

// ---------- O = bf16(gelu(w * G * K + Q)) ----------
__global__ void fuse_out(const float* __restrict__ wsm, const float* __restrict__ G,
                         const float* __restrict__ K, const float* __restrict__ Q,
                         unsigned short* __restrict__ O) {
    size_t i = (size_t)blockIdx.x * 256 + threadIdx.x;
    int d = (int)(i & 511);
    size_t row = i >> 9;
    int b = (int)(row / NN);
    float val = wsm[row] * G[(b << 9) + d] * K[i] + Q[i];
    O[i] = f2bf(gelu_exact(val));
}

extern "C" void kernel_launch(void* const* d_in, const int* in_sizes, int n_in,
                              void* d_out, int out_size, void* d_ws, size_t ws_size,
                              hipStream_t stream) {
    const float* x    = (const float*)d_in[0];
    const float* wq   = (const float*)d_in[1];
    const float* bq   = (const float*)d_in[2];
    const float* wk   = (const float*)d_in[3];
    const float* bk   = (const float*)d_in[4];
    const float* w_g  = (const float*)d_in[5];
    const float* wd1  = (const float*)d_in[6];
    const float* bd1  = (const float*)d_in[7];
    const float* wd2  = (const float*)d_in[8];
    const float* bd2  = (const float*)d_in[9];
    const float* wf   = (const float*)d_in[10];
    const float* bfb  = (const float*)d_in[11];

    char* ws = (char*)d_ws;
    size_t off = 0;
    auto take = [&](size_t bytes) -> char* {
        char* p = ws + off;
        off += (bytes + 255) & ~(size_t)255;
        return p;
    };
    unsigned short* Xbf  = (unsigned short*)take((size_t)RT * 512 * 2); // X -> H -> O (reused)
    unsigned short* Qbf  = (unsigned short*)take((size_t)RT * 512 * 2);
    float*          Qf   = (float*)take((size_t)RT * 512 * 4);
    float*          Kf   = (float*)take((size_t)RT * 512 * 4);
    unsigned short* WqT  = (unsigned short*)take(512 * 512 * 2);
    unsigned short* WkT  = (unsigned short*)take(512 * 512 * 2);
    unsigned short* Wd1T = (unsigned short*)take(512 * 512 * 2);
    unsigned short* WfT  = (unsigned short*)take(512 * 256 * 2);
    float*          qg   = (float*)take((size_t)BB * NN * 4);
    float*          rn   = (float*)take(64);
    float*          Gb   = (float*)take(BB * 512 * 4);
    float*          dyn  = (float*)take((size_t)BB * NN * 4);
    float*          wsm  = (float*)take((size_t)BB * NN * 4);

    dim3 tb(32, 8, 1);
    // x [B,512,3136] -> Xbf [B,3136,512] bf16; weights -> transposed bf16
    transpose_f32_bf16<<<dim3(98, 16, BB), tb, 0, stream>>>(x, Xbf, 512, NN,
                                                            (size_t)512 * NN, (size_t)NN * 512);
    transpose_f32_bf16<<<dim3(16, 16, 1), tb, 0, stream>>>(wq,  WqT,  512, 512, 0, 0);
    transpose_f32_bf16<<<dim3(16, 16, 1), tb, 0, stream>>>(wk,  WkT,  512, 512, 0, 0);
    transpose_f32_bf16<<<dim3(16, 16, 1), tb, 0, stream>>>(wd1, Wd1T, 512, 512, 0, 0);
    transpose_f32_bf16<<<dim3(8,  16, 1), tb, 0, stream>>>(wf,  WfT,  512, 256, 0, 0);

    // Q = X@Wq + bq ; K = X@Wk + bk   (fp32 out)
    // supertiles: (50176/32)*(512/32) = 1568*16 = 25088 -> 3136 blocks of 8 waves
    gemm16<<<3136, 256, 0, stream>>>(Xbf, WqT, bq, Qf, nullptr, RT, 512, 0);
    gemm16<<<3136, 256, 0, stream>>>(Xbf, WkT, bk, Kf, nullptr, RT, 512, 0);

    // row l2-norms; Q also -> bf16 and qg = (Q . w_g) * 256^-0.5
    normalize_rows<<<6272, 256, 0, stream>>>(Qf, Qbf, w_g, qg, RT, 3);
    normalize_rows<<<6272, 256, 0, stream>>>(Kf, nullptr, nullptr, nullptr, RT, 0);

    // A-normalization factor and global query G
    batch_rnorm<<<BB, 256, 0, stream>>>(qg, rn);
    global_query<<<dim3(BB, 2), 256, 0, stream>>>(qg, rn, Qf, Gb);

    // H = bf16(gelu(Q@Wd1 + bd1))  (reuses Xbf storage)
    gemm16<<<3136, 256, 0, stream>>>(Qbf, Wd1T, bd1, nullptr, Xbf, RT, 512, 1);

    // dyn = H . wd2 + bd2 ; softmax over tokens
    dyn_rowdot<<<6272, 256, 0, stream>>>(Xbf, wd2, bd2, dyn, RT);
    softmax_rows<<<BB, 256, 0, stream>>>(dyn, wsm);

    // O = bf16(gelu(w*G*K + Q))  (reuses Xbf storage)
    fuse_out<<<100352, 256, 0, stream>>>(wsm, Gb, Kf, Qf, Xbf);

    // out[b,c,h,w] = (O @ Wf + bf), computed transposed so stores are contiguous in n
    // supertiles: (256/32)*(50176/32) = 8*1568 = 12544 -> 1568 blocks
    gemm16<<<1568, 256, 0, stream>>>(WfT, Xbf, bfb, (float*)d_out, nullptr, TD, RT, 2);
}
